// Head_65541200937353
// MI455X (gfx1250) — matrix operand order repaired and checked by reading, not compiled, not run on hardware
//
#include <hip/hip_runtime.h>
#include <cmath>

// MI455X / gfx1250 fused RoPE-attention.
//   prep:      Wq/Wk/Wv -> f16 transposed [n][k]; RoPE cos/sin table (fp32).
//   qkv_rope:  q,k,v = x@W via v_wmma_f32_16x16x32_f16, x loaded once for all
//              three matrices; RoPE + 1/sqrt(hs) applied in-register.
//   attn:      per-batch flash-style attention entirely from LDS with WMMA.

typedef __attribute__((ext_vector_type(16))) _Float16 v16h;
typedef __attribute__((ext_vector_type(8)))  _Float16 v8h;
typedef __attribute__((ext_vector_type(8)))  float    v8f;

#define N_EMBD   1024
#define TLEN     128
#define HS       128
#define LDS_PITCH 136   // 128 + 8 halfs padding: conflict-free ds_load_b128

// ---- fragment loaders (CDNA5 wave32 WMMA f16 layouts) ----------------------
// A 16x32: lane row = lane%16; halfs 0..7 -> K=kbA..kbA+7, 8..15 -> +16
__device__ __forceinline__ v16h ldfragA(const _Float16* p) {
  v8h lo = *(const v8h*)p;
  v8h hi = *(const v8h*)(p + 16);
  return __builtin_shufflevector(lo, hi, 0,1,2,3,4,5,6,7,8,9,10,11,12,13,14,15);
}
// B 32x16 (column-major per lane): lane col = lane%16; halfs 0..7 -> K=kbB..kbB+7, 8..15 -> +16
__device__ __forceinline__ v16h ldfragB(const _Float16* p) {
  v8h lo = *(const v8h*)p;
  v8h hi = *(const v8h*)(p + 16);
  return __builtin_shufflevector(lo, hi, 0,1,2,3,4,5,6,7,8,9,10,11,12,13,14,15);
}

// ---- kernel 0: weight transpose (f32 -> f16 [m][n][k]) + rope table --------
__global__ __launch_bounds__(256) void prep_kernel(
    const float* __restrict__ Wq, const float* __restrict__ Wk,
    const float* __restrict__ Wv, _Float16* __restrict__ wT,
    float2* __restrict__ rope)
{
  int idx = blockIdx.x * 256 + threadIdx.x;          // 0 .. 3*128*1024-1
  int m   = idx >> 17;
  int rem = idx & 131071;
  int n   = rem >> 10;
  int kk  = rem & 1023;
  const float* W = (m == 0) ? Wq : ((m == 1) ? Wk : Wv);
  wT[idx] = (_Float16)W[kk * HS + n];
  if (idx < TLEN * (HS / 2)) {                       // 128*64 rope entries
    int t = idx >> 6, i = idx & 63;
    // inv_freq = exp(-ln(10000)/64 * i)
    float ang = (float)t * expf(-0.14391156831212787f * (float)i);
    rope[idx] = make_float2(cosf(ang), sinf(ang));
  }
}

// ---- kernel 1: q,k,v GEMM + RoPE -------------------------------------------
__global__ __launch_bounds__(256) void qkv_rope_kernel(
    const float* __restrict__ x, const _Float16* __restrict__ wT,
    const float2* __restrict__ rope,
    _Float16* __restrict__ qo, _Float16* __restrict__ ko,
    _Float16* __restrict__ vo)
{
  const int lane = threadIdx.x & 31;
  const int wv   = threadIdx.x >> 5;          // 8 waves
  const int r    = lane & 15;
  const int hl   = lane >> 4;
  const int kbA  = hl * 8;
  const int kbB  = hl * 8;
  const int rowbase = blockIdx.x * 128 + wv * 16;   // 16-row strip per wave

  const float* xrow = x + (size_t)(rowbase + r) * N_EMBD + kbA;

  v8f acc[3][8] = {};                          // q/k/v x 8 N-tiles (N=128)

  #pragma unroll 1
  for (int kk = 0; kk < N_EMBD; kk += 32) {
    const float* xp = xrow + kk;
    float4 f0 = *(const float4*)(xp);
    float4 f1 = *(const float4*)(xp + 4);
    float4 f2 = *(const float4*)(xp + 16);
    float4 f3 = *(const float4*)(xp + 20);
    v16h a;
    a[0]=(_Float16)f0.x;  a[1]=(_Float16)f0.y;  a[2]=(_Float16)f0.z;  a[3]=(_Float16)f0.w;
    a[4]=(_Float16)f1.x;  a[5]=(_Float16)f1.y;  a[6]=(_Float16)f1.z;  a[7]=(_Float16)f1.w;
    a[8]=(_Float16)f2.x;  a[9]=(_Float16)f2.y;  a[10]=(_Float16)f2.z; a[11]=(_Float16)f2.w;
    a[12]=(_Float16)f3.x; a[13]=(_Float16)f3.y; a[14]=(_Float16)f3.z; a[15]=(_Float16)f3.w;
    #pragma unroll
    for (int m = 0; m < 3; ++m) {
      const _Float16* wbase = wT + m * (N_EMBD * HS) + kk + kbB;
      #pragma unroll
      for (int nt = 0; nt < 8; ++nt) {
        v16h b = ldfragB(wbase + (nt * 16 + r) * N_EMBD);
        acc[m][nt] = __builtin_amdgcn_wmma_f32_16x16x32_f16(
            false, a, false, b, (short)0, acc[m][nt], false, false);
      }
    }
  }

  // Epilogue: RoPE on q,k (pairs live in adjacent lanes of C layout), store f16
  const int tbase = rowbase & (TLEN - 1);
  #pragma unroll
  for (int g = 0; g < 8; ++g) {
    const int t    = tbase + g + 8 * hl;
    const size_t grow = (size_t)(rowbase + g + 8 * hl);
    #pragma unroll
    for (int nt = 0; nt < 8; ++nt) {
      const int n = nt * 16 + r;
      const float2 cs = rope[t * 64 + (n >> 1)];
      float vq = acc[0][nt][g];
      float pq = __shfl_xor(vq, 1);
      float oq = (n & 1) ? (pq * cs.y + vq * cs.x) : (vq * cs.x - pq * cs.y);
      qo[grow * HS + n] = (_Float16)(oq * 0.08838834764831845f); // 1/sqrt(128)
      float vk = acc[1][nt][g];
      float pk = __shfl_xor(vk, 1);
      float ok = (n & 1) ? (pk * cs.y + vk * cs.x) : (vk * cs.x - pk * cs.y);
      ko[grow * HS + n] = (_Float16)ok;
      vo[grow * HS + n] = (_Float16)acc[2][nt][g];
    }
  }
}

// ---- kernel 2: causal softmax attention, one workgroup per batch -----------
__global__ __launch_bounds__(256) void attn_kernel(
    const _Float16* __restrict__ q, const _Float16* __restrict__ k,
    const _Float16* __restrict__ v, float* __restrict__ out)
{
  extern __shared__ _Float16 smem[];
  _Float16* sQ = smem;                         // [t][h], reused as P later
  _Float16* sK = smem + TLEN * LDS_PITCH;      // [t][h]
  _Float16* sV = smem + 2 * TLEN * LDS_PITCH;  // transposed [h][t]

  const int b   = blockIdx.x;
  const int tid = threadIdx.x;
  const size_t base = (size_t)b * (TLEN * HS);

  #pragma unroll
  for (int it = 0; it < 8; ++it) {
    int i = (tid + it * 256) * 8;
    int t = i >> 7, h = i & 127;
    *(v8h*)(sQ + t * LDS_PITCH + h) = *(const v8h*)(q + base + i);
    *(v8h*)(sK + t * LDS_PITCH + h) = *(const v8h*)(k + base + i);
    v8h d = *(const v8h*)(v + base + i);
    #pragma unroll
    for (int j = 0; j < 8; ++j) sV[(h + j) * LDS_PITCH + t] = d[j];
  }
  __syncthreads();

  const int lane = tid & 31, wv = tid >> 5, r = lane & 15, hl = lane >> 4;
  const int kbA = hl * 8, kbB = hl * 8;

  // S = Q K^T  (wave wv owns query rows 16wv..16wv+15; all 8 key tiles)
  v8f s[8] = {};
  #pragma unroll
  for (int kk = 0; kk < HS; kk += 32) {
    v16h a = ldfragA(sQ + (wv * 16 + r) * LDS_PITCH + kk + kbA);
    #pragma unroll
    for (int ct = 0; ct < 8; ++ct) {
      v16h bb = ldfragB(sK + (ct * 16 + r) * LDS_PITCH + kk + kbB);
      s[ct] = __builtin_amdgcn_wmma_f32_16x16x32_f16(
          false, a, false, bb, (short)0, s[ct], false, false);
    }
  }

  // causal mask: key > query  ->  -inf
  #pragma unroll
  for (int ct = 0; ct < 8; ++ct)
    #pragma unroll
    for (int g = 0; g < 8; ++g)
      if (ct * 16 + r > wv * 16 + g + 8 * hl) s[ct][g] = -__builtin_inff();

  // row softmax (row = (g, lane-half); xor masks 1/2/4/8 stay in the half)
  float rinv[8];
  #pragma unroll
  for (int g = 0; g < 8; ++g) {
    float m = -__builtin_inff();
    #pragma unroll
    for (int ct = 0; ct < 8; ++ct) m = fmaxf(m, s[ct][g]);
    m = fmaxf(m, __shfl_xor(m, 1));
    m = fmaxf(m, __shfl_xor(m, 2));
    m = fmaxf(m, __shfl_xor(m, 4));
    m = fmaxf(m, __shfl_xor(m, 8));
    float sum = 0.f;
    #pragma unroll
    for (int ct = 0; ct < 8; ++ct) {
      float e = __expf(s[ct][g] - m);
      s[ct][g] = e;
      sum += e;
    }
    sum += __shfl_xor(sum, 1);
    sum += __shfl_xor(sum, 2);
    sum += __shfl_xor(sum, 4);
    sum += __shfl_xor(sum, 8);
    rinv[g] = 1.f / sum;
  }

  // P -> LDS (wave-private rows of sQ; LDS ops are in-order per wave)
  #pragma unroll
  for (int ct = 0; ct < 8; ++ct)
    #pragma unroll
    for (int g = 0; g < 8; ++g)
      sQ[(wv * 16 + g + 8 * hl) * LDS_PITCH + ct * 16 + r] = (_Float16)s[ct][g];

  // O = P V
  v8f o[8] = {};
  #pragma unroll
  for (int kk = 0; kk < TLEN; kk += 32) {
    v16h a = ldfragA(sQ + (wv * 16 + r) * LDS_PITCH + kk + kbA);
    #pragma unroll
    for (int nt = 0; nt < 8; ++nt) {
      v16h bb = ldfragB(sV + (nt * 16 + r) * LDS_PITCH + kk + kbB);
      o[nt] = __builtin_amdgcn_wmma_f32_16x16x32_f16(
          false, a, false, bb, (short)0, o[nt], false, false);
    }
  }

  #pragma unroll
  for (int g = 0; g < 8; ++g) {
    const size_t row = (size_t)b * TLEN + wv * 16 + g + 8 * hl;
    #pragma unroll
    for (int nt = 0; nt < 8; ++nt)
      out[row * HS + nt * 16 + r] = o[nt][g] * rinv[g];
  }
}

// ---- launcher --------------------------------------------------------------
extern "C" void kernel_launch(void* const* d_in, const int* in_sizes, int n_in,
                              void* d_out, int out_size, void* d_ws, size_t ws_size,
                              hipStream_t stream) {
  (void)in_sizes; (void)n_in; (void)out_size; (void)ws_size;
  const float* x  = (const float*)d_in[0];
  const float* Wq = (const float*)d_in[1];
  const float* Wk = (const float*)d_in[2];
  const float* Wv = (const float*)d_in[3];

  char* ws = (char*)d_ws;                       // ~26.3 MB used
  _Float16* qws = (_Float16*)(ws);              //  8 MB
  _Float16* kws = (_Float16*)(ws + (size_t)(8u << 20));
  _Float16* vws = (_Float16*)(ws + (size_t)(16u << 20));
  _Float16* wT  = (_Float16*)(ws + (size_t)(24u << 20));          // 768 KB
  float2*   rope = (float2*)(ws + (size_t)(24u << 20) + (1u << 20)); // 64 KB

  prep_kernel<<<1536, 256, 0, stream>>>(Wq, Wk, Wv, wT, rope);
  qkv_rope_kernel<<<256, 256, 0, stream>>>(x, wT, rope, qws, kws, vws);
  attn_kernel<<<256, 256, 3 * TLEN * LDS_PITCH * sizeof(_Float16), stream>>>(
      qws, kws, vws, (float*)d_out);
}
